// LSTMDecoder_68478958567587
// MI455X (gfx1250) — matrix operand order repaired and checked by reading, not compiled, not run on hardware
//
#include <hip/hip_runtime.h>
#include <stdint.h>

// ---------------------------------------------------------------------------
// CDNA5 (gfx1250) LSTM decoder.  wave32, WMMA f32_16x16x32_bf16 everywhere.
// ---------------------------------------------------------------------------

typedef __attribute__((ext_vector_type(16))) __bf16 v16bf;
typedef __attribute__((ext_vector_type(8)))  float  v8f;

struct Q32 { uint4 a, b; };   // 32 bytes = 16 bf16 = one WMMA A/B fragment per lane
struct H8  { __bf16 h[8]; };  // 16 bytes = 8 bf16
struct F8  { float  f[8]; };  // 32 bytes = 8 f32

__device__ __forceinline__ v16bf frag16(const __bf16* p0, const __bf16* p1) {
  Q32 q;
  q.a = *reinterpret_cast<const uint4*>(p0);
  q.b = *reinterpret_cast<const uint4*>(p1);
  return __builtin_bit_cast(v16bf, q);
}

__device__ __forceinline__ F8 ldF8(const float* p) {
  struct { float4 a, b; } q;
  q.a = *reinterpret_cast<const float4*>(p);
  q.b = *reinterpret_cast<const float4*>(p + 4);
  return __builtin_bit_cast(F8, q);
}

__device__ __forceinline__ float sigm(float x) { return 1.0f / (1.0f + __expf(-x)); }

// ---------------------------------------------------------------------------
// Elementwise helpers
// ---------------------------------------------------------------------------
__global__ void k_embed(const int* __restrict__ ids, const float* __restrict__ emb,
                        __bf16* __restrict__ x, int total) {
  int i = blockIdx.x * blockDim.x + threadIdx.x;
  if (i >= total) return;
  int e  = i & 255;          // E = 256
  int bt = i >> 8;
  x[i] = (__bf16)emb[(size_t)ids[bt] * 256 + e];
}

__global__ void k_f2bf(const float* __restrict__ s, __bf16* __restrict__ d, int n) {
  int i = blockIdx.x * blockDim.x + threadIdx.x;
  if (i < n) d[i] = (__bf16)s[i];
}

__global__ void k_biassum(const float* __restrict__ a, const float* __restrict__ b,
                          float* __restrict__ d, int n) {
  int i = blockIdx.x * blockDim.x + threadIdx.x;
  if (i < n) d[i] = a[i] + b[i];
}

// Pack W_hh (l, 4H=1024 rows, H=256 cols, f32) into this file's per-lane B-fragment
// order for v_wmma_f32_16x16x32_bf16 (same k order as the A fragments): for each (kstep, ntile), lane (lg=lane>>4,
// lr=lane&15) holds N = ntile*16+lr, K = kstep*32 + lg*16 + {0..15} sequential.
__global__ void k_pack_whh(const float* __restrict__ W, __bf16* __restrict__ P) {
  int tid   = blockIdx.x * blockDim.x + threadIdx.x;   // L*8*64*32 = 49152 threads
  int lane  = tid & 31;
  int ntile = (tid >> 5) & 63;
  int kstep = (tid >> 11) & 7;
  int l     = tid >> 14;
  if (l >= 3) return;
  int lr = lane & 15, lg = lane >> 4;
  const float* src = W + (size_t)l * 1024 * 256 + (size_t)(ntile * 16 + lr) * 256
                       + kstep * 32 + lg * 16;
  __bf16* dst = P + (size_t)l * 1024 * 256 + ((((size_t)kstep * 64 + ntile) * 32 + lane) << 4);
#pragma unroll
  for (int j = 0; j < 16; ++j) dst[j] = (__bf16)src[j];
}

// ---------------------------------------------------------------------------
// Generic WMMA GEMM:  C(M,N) = A(M,K) * B(N,K)^T + bias
// A,B bf16 row-major.  One wave computes a 16(M) x 64(N) tile (A-frag reuse),
// software-pipelined: fragments for k+1 are in flight while k's WMMAs issue.
// mode 0: f32 out[m*N+n]   mode 1: bf16 out[m*N+n]
// mode 2: token head: m=(b,t), skip t==Tseq-1, f32 out[(b*(Tseq-1)+t)*N+n]
// ---------------------------------------------------------------------------
__global__ void __launch_bounds__(256)
k_gemm(const __bf16* __restrict__ A, int lda,
       const __bf16* __restrict__ Bw,
       const float* __restrict__ bias,
       int M, int N, int K, int mode, void* __restrict__ out, int Tseq) {
  const int wave = blockIdx.x * (blockDim.x >> 5) + (threadIdx.x >> 5);
  const int lane = threadIdx.x & 31;
  const int ngrp = N >> 6;
  const int mt = wave / ngrp;
  const int ng = wave % ngrp;
  if (mt * 16 >= M) return;                    // wave-uniform
  const int lg = lane >> 4, lr = lane & 15;
  const int kiters = K >> 5;

  const __bf16* arow = A + (size_t)(mt * 16 + lr) * lda + lg * 16;
  const __bf16* brow0 = Bw + (size_t)((ng << 6) + lr) * K + lg * 16;

  v8f acc[4] = {};
  v16bf a0 = frag16(arow, arow + 8);
  v16bf b0[4], b1[4];
#pragma unroll
  for (int nt = 0; nt < 4; ++nt) {
    const __bf16* p = brow0 + (size_t)nt * 16 * K;
    b0[nt] = frag16(p, p + 8);
  }
  for (int k = 0; k < kiters; ++k) {
    const int kn = (k + 1 < kiters) ? (k + 1) : k;   // last iter: benign reload
    const __bf16* an = arow + (kn << 5);
    v16bf a1 = frag16(an, an + 8);
#pragma unroll
    for (int nt = 0; nt < 4; ++nt) {
      const __bf16* p = brow0 + (size_t)nt * 16 * K + (kn << 5);
      b1[nt] = frag16(p, p + 8);
    }
#pragma unroll
    for (int nt = 0; nt < 4; ++nt)
      acc[nt] = __builtin_amdgcn_wmma_f32_16x16x32_bf16(
          false, a0, false, b0[nt], (short)0, acc[nt], false, false);
    a0 = a1;
#pragma unroll
    for (int nt = 0; nt < 4; ++nt) b0[nt] = b1[nt];
  }

#pragma unroll
  for (int nt = 0; nt < 4; ++nt) {
    const int n = (ng << 6) + nt * 16 + lr;
    const float bv = bias ? bias[n] : 0.0f;
#pragma unroll
    for (int vv = 0; vv < 8; ++vv) {
      const int m = (mt << 4) + vv + (lg << 3);
      const float val = acc[nt][vv] + bv;
      if (mode == 0) {
        ((float*)out)[(size_t)m * N + n] = val;
      } else if (mode == 1) {
        ((__bf16*)out)[(size_t)m * N + n] = (__bf16)val;
      } else {
        const int t = m % Tseq, b = m / Tseq;
        if (t < Tseq - 1)
          ((float*)out)[((size_t)b * (Tseq - 1) + t) * N + n] = val;
      }
    }
  }
}

// ---------------------------------------------------------------------------
// Recurrent scan.  One workgroup owns 16 batch rows for all T steps.
// Per step: gates = xg[b,t,:] (bias folded) + h @ W_hh^T  -> LSTM update.
// h kept in LDS (bf16, padded stride), c in registers, gates via LDS f32.
// 16 waves: wave w computes gate columns [w*64, w*64+64).
// W_hh fragments for K-steps 0..3 are cached in VGPRs across all T steps
// (128 VGPRs/lane); K-steps 4..7 stream from L2 each step.
// ---------------------------------------------------------------------------
#define HSTRIDE 264
__global__ void __launch_bounds__(512)
k_scan(const __bf16* __restrict__ xg,     // (B,T,4H) bf16, biases folded in
       const __bf16* __restrict__ Wp,     // packed W_hh fragments
       const float* __restrict__ h0, const float* __restrict__ c0,
       __bf16* __restrict__ xout,         // (B,T,H) bf16
       float* __restrict__ hT,            // (B,H) f32
       int T) {
  __shared__ float  gates[16 * 1024];
  __shared__ __bf16 hsh[16 * HSTRIDE];

  const int tid  = threadIdx.x;
  const int w    = tid >> 5;            // wave id == batch-local row for elementwise
  const int lane = tid & 31;
  const int lg = lane >> 4, lr = lane & 15;
  const int bbase = blockIdx.x * 16;
  const int eh0 = (tid & 31) * 8;       // 8 hidden units per thread

  // Resident W_hh fragments for K-steps 0..3 (loaded once, reused all T steps)
  v16bf bc[4][4];
#pragma unroll
  for (int k = 0; k < 4; ++k)
#pragma unroll
    for (int nt = 0; nt < 4; ++nt) {
      const __bf16* p = Wp + ((((size_t)k * 64 + (w * 4 + nt)) * 32 + lane) << 4);
      bc[k][nt] = frag16(p, p + 8);
    }

  float c[8];
#pragma unroll
  for (int j = 0; j < 8; ++j) {
    const int gi = (bbase + w) * 256 + eh0 + j;
    c[j] = c0[gi];
    hsh[w * HSTRIDE + eh0 + j] = (__bf16)h0[gi];
  }

  for (int t = 0; t < T; ++t) {
    __syncthreads();                    // h ready, gates free
    const size_t xgb = ((size_t)(bbase + w) * T + t) * 1024 + eh0;
    // Pull this step's gate inputs toward the caches under the GEMM's shadow
    __builtin_prefetch((const void*)(xg + xgb), 0, 0);
    __builtin_prefetch((const void*)(xg + xgb + 256), 0, 0);
    __builtin_prefetch((const void*)(xg + xgb + 512), 0, 0);
    __builtin_prefetch((const void*)(xg + xgb + 768), 0, 0);

    // ---- GEMM: h(16x256) @ W_hh^T slice -> gates columns [w*64, w*64+64)
    v8f acc[4] = {};
#pragma unroll
    for (int k = 0; k < 4; ++k) {       // cached-weight half
      const __bf16* ap = &hsh[lr * HSTRIDE + (k << 5) + lg * 16];
      v16bf af = frag16(ap, ap + 8);
#pragma unroll
      for (int nt = 0; nt < 4; ++nt)
        acc[nt] = __builtin_amdgcn_wmma_f32_16x16x32_bf16(
            false, af, false, bc[k][nt], (short)0, acc[nt], false, false);
    }
#pragma unroll
    for (int k = 4; k < 8; ++k) {       // streamed-weight half
      const __bf16* ap = &hsh[lr * HSTRIDE + (k << 5) + lg * 16];
      v16bf af = frag16(ap, ap + 8);
      v16bf bs_[4];
#pragma unroll
      for (int nt = 0; nt < 4; ++nt) {
        const __bf16* p = Wp + ((((size_t)k * 64 + (w * 4 + nt)) * 32 + lane) << 4);
        bs_[nt] = frag16(p, p + 8);
      }
#pragma unroll
      for (int nt = 0; nt < 4; ++nt)
        acc[nt] = __builtin_amdgcn_wmma_f32_16x16x32_bf16(
            false, af, false, bs_[nt], (short)0, acc[nt], false, false);
    }
#pragma unroll
    for (int nt = 0; nt < 4; ++nt)
#pragma unroll
      for (int vv = 0; vv < 8; ++vv)
        gates[(vv + (lg << 3)) * 1024 + (w << 6) + nt * 16 + lr] = acc[nt][vv];
    __syncthreads();                    // gates ready, h reads done

    // ---- LSTM elementwise: thread -> (row w, hidden eh0..eh0+7), vector ops
    const float* grow = &gates[w * 1024 + eh0];
    const F8 gi8 = ldF8(grow);
    const F8 gf8 = ldF8(grow + 256);
    const F8 gg8 = ldF8(grow + 512);
    const F8 go8 = ldF8(grow + 768);
    const H8 xi8 = __builtin_bit_cast(H8, *(const uint4*)(xg + xgb));
    const H8 xf8 = __builtin_bit_cast(H8, *(const uint4*)(xg + xgb + 256));
    const H8 xg8 = __builtin_bit_cast(H8, *(const uint4*)(xg + xgb + 512));
    const H8 xo8 = __builtin_bit_cast(H8, *(const uint4*)(xg + xgb + 768));
    H8 ho;
#pragma unroll
    for (int j = 0; j < 8; ++j) {
      const float fi = sigm(gi8.f[j] + (float)xi8.h[j]);
      const float ff = sigm(gf8.f[j] + (float)xf8.h[j]);
      const float fo = sigm(go8.f[j] + (float)xo8.h[j]);
      c[j] = ff * c[j] + fi * tanhf(gg8.f[j] + (float)xg8.h[j]);
      ho.h[j] = (__bf16)(fo * tanhf(c[j]));
    }
    *reinterpret_cast<uint4*>(&hsh[w * HSTRIDE + eh0]) = __builtin_bit_cast(uint4, ho);
    const size_t xob = ((size_t)(bbase + w) * T + t) * 256 + eh0;
    *reinterpret_cast<uint4*>(xout + xob) = __builtin_bit_cast(uint4, ho);
    if (t == T - 1) {
#pragma unroll
      for (int j = 0; j < 8; ++j)
        hT[(bbase + w) * 256 + eh0 + j] = (float)ho.h[j];
    }
  }
}

// ---------------------------------------------------------------------------
// Atom head: relu(hT @ a1_W^T + a1_b) @ a2_W^T + a2_b   (tiny, f32 VALU)
// ---------------------------------------------------------------------------
__global__ void k_atom(const float* __restrict__ hT,
                       const float* __restrict__ a1W, const float* __restrict__ a1b,
                       const float* __restrict__ a2W, const float* __restrict__ a2b,
                       float* __restrict__ out) {
  __shared__ float s[256];
  const int b = blockIdx.x, n = threadIdx.x;
  const float* h = hT + (size_t)b * 256;
  float acc = a1b[n];
  for (int k = 0; k < 256; ++k) acc = fmaf(h[k], a1W[n * 256 + k], acc);
  s[n] = fmaxf(acc, 0.0f);
  __syncthreads();
  if (n < 9) {
    float a = a2b[n];
    for (int k = 0; k < 256; ++k) a = fmaf(s[k], a2W[n * 256 + k], a);
    out[(size_t)b * 9 + n] = a;
  }
}

// ---------------------------------------------------------------------------
extern "C" void kernel_launch(void* const* d_in, const int* in_sizes, int n_in,
                              void* d_out, int out_size, void* d_ws, size_t ws_size,
                              hipStream_t stream) {
  (void)in_sizes; (void)n_in; (void)out_size; (void)ws_size;
  const int*   ids   = (const int*)  d_in[0];
  const float* cond  = (const float*)d_in[1];
  const float* emb   = (const float*)d_in[2];
  const float* h0W   = (const float*)d_in[3];
  const float* h0bi  = (const float*)d_in[4];
  const float* c0W   = (const float*)d_in[5];
  const float* c0bi  = (const float*)d_in[6];
  const float* Wih   = (const float*)d_in[7];
  const float* Whh   = (const float*)d_in[8];
  const float* bih   = (const float*)d_in[9];
  const float* bhh   = (const float*)d_in[10];
  const float* tokW  = (const float*)d_in[11];
  const float* tokb  = (const float*)d_in[12];
  const float* a1W   = (const float*)d_in[13];
  const float* a1b   = (const float*)d_in[14];
  const float* a2W   = (const float*)d_in[15];
  const float* a2b   = (const float*)d_in[16];

  const int Bsz = 256, T = 512, V = 128;
  const size_t BT = (size_t)Bsz * T;           // 131072

  char* wp = (char*)d_ws;
  auto carve = [&](size_t bytes) -> char* {
    char* p = wp; wp += (bytes + 255) & ~(size_t)255; return p;
  };
  __bf16* xA      = (__bf16*)carve(BT * 256 * 2);        // 64 MiB
  __bf16* xB      = (__bf16*)carve(BT * 256 * 2);        // 64 MiB
  __bf16* xg      = (__bf16*)carve(BT * 1024 * 2);       // 256 MiB
  __bf16* cond_bf = (__bf16*)carve(262144 * 2);
  __bf16* h0W_bf  = (__bf16*)carve(262144 * 2);
  __bf16* c0W_bf  = (__bf16*)carve(262144 * 2);
  __bf16* Wih_bf  = (__bf16*)carve(786432 * 2);
  __bf16* Whh_pk  = (__bf16*)carve(786432 * 2);
  __bf16* tokW_bf = (__bf16*)carve(32768 * 2);
  float*  bs      = (float*) carve(3072 * 4);
  float*  h0buf   = (float*) carve(65536 * 4);
  float*  c0buf   = (float*) carve(65536 * 4);
  float*  hTbuf   = (float*) carve(65536 * 4);

  // Preprocessing (parallel, independent)
  k_embed  <<<131072, 256, 0, stream>>>(ids, emb, xA, (int)(BT * 256));
  k_f2bf   <<<1024,  256, 0, stream>>>(cond, cond_bf, 262144);
  k_f2bf   <<<1024,  256, 0, stream>>>(h0W,  h0W_bf,  262144);
  k_f2bf   <<<1024,  256, 0, stream>>>(c0W,  c0W_bf,  262144);
  k_f2bf   <<<3072,  256, 0, stream>>>(Wih,  Wih_bf,  786432);
  k_f2bf   <<<128,   256, 0, stream>>>(tokW, tokW_bf, 32768);
  k_pack_whh<<<192,  256, 0, stream>>>(Whh, Whh_pk);
  k_biassum<<<12,    256, 0, stream>>>(bih, bhh, bs, 3072);

  // h0 / c0 projections: (256 x 1024) @ (256 x 1024)^T  -> f32
  k_gemm<<<8, 256, 0, stream>>>(cond_bf, 1024, h0W_bf, h0bi, 256, 256, 1024, 0, h0buf, 0);
  k_gemm<<<8, 256, 0, stream>>>(cond_bf, 1024, c0W_bf, c0bi, 256, 256, 1024, 0, c0buf, 0);

  // 3 LSTM layers: big parallel input GEMM + persistent-tile recurrent scan
  __bf16* cur_in = xA;
  __bf16* cur_out = xB;
  for (int l = 0; l < 3; ++l) {
    k_gemm<<<16384, 256, 0, stream>>>(cur_in, 256, Wih_bf + (size_t)l * 262144,
                                      bs + l * 1024, (int)BT, 1024, 256, 1, xg, 0);
    k_scan<<<16, 512, 0, stream>>>(xg, Whh_pk + (size_t)l * 262144,
                                   h0buf, c0buf, cur_out, hTbuf, T);
    __bf16* tmp = cur_in; cur_in = cur_out; cur_out = tmp;
  }

  // token logits: x3[:, :-1] @ tok_W^T + tok_b  -> d_out[0 .. B*(T-1)*V)
  k_gemm<<<2048, 256, 0, stream>>>(cur_in, 256, tokW_bf, tokb,
                                   (int)BT, V, 256, 2, d_out, T);
  // atom head -> d_out tail
  k_atom<<<256, 256, 0, stream>>>(hTbuf, a1W, a1b, a2W, a2b,
                                  (float*)d_out + (size_t)Bsz * (T - 1) * V);
}
